// DGCNNReg_14817637171207
// MI455X (gfx1250) — compile-verified, matrix operand order checked by reading, not run on hardware
//
#include <hip/hip_runtime.h>
#include <cstdint>

typedef __attribute__((ext_vector_type(16))) _Float16 v16h;
typedef __attribute__((ext_vector_type(8)))  _Float16 v8h;
typedef __attribute__((ext_vector_type(8)))  float    v8f;

#define KNN   20
#define SLOPE 0.2f
#define BM    128
#define BN    64
#define BK    32
#define LDSLD 40   // halves per LDS row (32 + 8 pad -> conflict-free b128 reads)

// ---------------------------------------------------------------------------
// Generic fused GEMM:  out = act( (A @ W) * scale[col] + shift[col] )
//   A  : M x Kd f16 row-major (lda)        (M multiple of 128, Kd multiple of 32)
//   Bt : N x Kd f16 row-major (= W^T, zero-padded)  (N multiple of 64)
//   scale/shift fold bias + BN:  y*s + (b - m)*s + beta
//   act: 0=none, 1=relu, 2=leaky(0.2)
//   optional f16 store (ld16) and f32 store (ld32, cols < ncols32)
// ---------------------------------------------------------------------------
__global__ __launch_bounds__(256, 2)
void wmma_gemm_kernel(const _Float16* __restrict__ A, int lda,
                      const _Float16* __restrict__ Bt,
                      const float* __restrict__ scale, const float* __restrict__ shift,
                      _Float16* __restrict__ out16, int ld16,
                      float* __restrict__ out32, int ld32, int ncols32,
                      int Kd, int act)
{
    __shared__ _Float16 As[BM * LDSLD];
    __shared__ _Float16 Bs[BN * LDSLD];

    const int tid  = threadIdx.x;
    const int wave = tid >> 5;
    const int lane = tid & 31;
    const int l16  = lane & 15;
    const int hi   = (lane >> 4) & 1;          // half-wave select
    const long long row0 = (long long)blockIdx.x * BM;
    const int col0 = blockIdx.y * BN;

    v8f acc[4] = {};

    for (int k0 = 0; k0 < Kd; k0 += BK) {
        // --- cooperative A tile load: 128 x 32 halves = 512 x 16B chunks ---
#pragma unroll
        for (int c = 0; c < 2; ++c) {
            int chunk = tid + c * 256;
            int r  = chunk >> 2;
            int co = (chunk & 3) << 3;
            const v8h* src = (const v8h*)(A + (row0 + r) * (long long)lda + k0 + co);
            *(v8h*)(&As[r * LDSLD + co]) = *src;
        }
        // --- B tile: 64 x 32 halves = 256 x 16B chunks (Bt is N x Kd) ---
        {
            int r  = tid >> 2;
            int co = (tid & 3) << 3;
            const v8h* src = (const v8h*)(Bt + (long long)(col0 + r) * Kd + k0 + co);
            *(v8h*)(&Bs[r * LDSLD + co]) = *src;
        }
        __syncthreads();

        // --- fragments per ISA 16-bit layout: lane<16 K{0-7,16-23}, lane>=16 K{8-15,24-31}
        const int kh = hi * 8;
        union { v16h v; v8h h[2]; } af;
        const _Float16* arow = &As[(wave * 16 + l16) * LDSLD];
        af.h[0] = *(const v8h*)(arow + kh);
        af.h[1] = *(const v8h*)(arow + kh + 16);
#pragma unroll
        for (int t = 0; t < 4; ++t) {
            union { v16h v; v8h h[2]; } bf;
            const _Float16* brow = &Bs[(t * 16 + l16) * LDSLD];
            bf.h[0] = *(const v8h*)(brow + kh);
            bf.h[1] = *(const v8h*)(brow + kh + 16);
            acc[t] = __builtin_amdgcn_wmma_f32_16x16x32_f16(
                         false, af.v, false, bf.v, (short)0, acc[t], false, false);
        }
        __syncthreads();
    }

    // --- epilogue: C/D layout VGPR r -> M = r + 8*hi, lane -> N = l16 ---
#pragma unroll
    for (int t = 0; t < 4; ++t) {
#pragma unroll
        for (int r = 0; r < 8; ++r) {
            long long row = row0 + wave * 16 + r + hi * 8;
            int col = col0 + t * 16 + l16;
            float y = acc[t][r] * scale[col] + shift[col];
            if (act == 1)      y = fmaxf(y, 0.f);
            else if (act == 2) y = (y > 0.f) ? y : SLOPE * y;
            if (out16) out16[row * (long long)ld16 + col] = (_Float16)y;
            if (out32 && col < ncols32) out32[row * (long long)ld32 + col] = y;
        }
    }
}

// ---------------------------------------------------------------------------
// Transpose x (N x 13) -> xT (13 x N) for coalesced kNN column loads
// ---------------------------------------------------------------------------
__global__ __launch_bounds__(256)
void xpose_kernel(const float* __restrict__ x, float* __restrict__ xT, int N)
{
    int tot = N * 13;
    for (int e = blockIdx.x * 256 + threadIdx.x; e < tot; e += gridDim.x * 256) {
        int c = e % 13, j = e / 13;
        xT[c * N + j] = x[e];
    }
}

// ---------------------------------------------------------------------------
// kNN top-20: one 128-thread block per point. Per-thread register top-20
// (branchless unrolled insert), then 20 rounds of block-wide argmin pops.
// ---------------------------------------------------------------------------
__global__ __launch_bounds__(128)
void knn_topk_kernel(const float* __restrict__ xT, const float* __restrict__ x,
                     int* __restrict__ idx, int N)
{
    const int i = blockIdx.x;
    const int t = threadIdx.x;
    __shared__ float sd[128];
    __shared__ int   sj[128];
    __shared__ int   st[128];

    float xi[13];
#pragma unroll
    for (int c = 0; c < 13; ++c) xi[c] = x[i * 13 + c];

    float bd[KNN]; int bi[KNN];
#pragma unroll
    for (int s = 0; s < KNN; ++s) { bd[s] = 1.0e38f; bi[s] = 0; }

    for (int j = t; j < N; j += 128) {
        float d = 0.f;
#pragma unroll
        for (int c = 0; c < 13; ++c) {
            float df = xT[c * N + j] - xi[c];
            d = fmaf(df, df, d);
        }
        if (j == i) d = 1.0e38f;
        if (d < bd[KNN - 1]) {
            float cd = d; int ci = j;
#pragma unroll
            for (int s = 0; s < KNN; ++s) {
                bool sw = cd < bd[s];
                float td = bd[s]; int ti = bi[s];
                if (sw) { bd[s] = cd; bi[s] = ci; cd = td; ci = ti; }
            }
        }
    }

    for (int r = 0; r < KNN; ++r) {
        sd[t] = bd[0]; sj[t] = bi[0]; st[t] = t;
        __syncthreads();
#pragma unroll
        for (int off = 64; off > 0; off >>= 1) {
            if (t < off && sd[t + off] < sd[t]) {
                sd[t] = sd[t + off]; sj[t] = sj[t + off]; st[t] = st[t + off];
            }
            __syncthreads();
        }
        if (t == 0) idx[i * KNN + r] = sj[0];
        int w = st[0];
        __syncthreads();
        if (t == w) {  // pop winner's head (static-index shift, no scratch)
#pragma unroll
            for (int s = 0; s < KNN - 1; ++s) { bd[s] = bd[s + 1]; bi[s] = bi[s + 1]; }
            bd[KNN - 1] = 1.0e38f;
        }
        __syncthreads();
    }
}

// ---------------------------------------------------------------------------
// Build edge messages [xi, xj-xi] in f16, K-padded with zeros to kpad
// ---------------------------------------------------------------------------
__global__ __launch_bounds__(256)
void gather_msg_kernel(const float* __restrict__ h, int ldh,
                       const int* __restrict__ idx,
                       _Float16* __restrict__ msg, int din, int kpad,
                       long long total)
{
    for (long long e = (long long)blockIdx.x * 256 + threadIdx.x; e < total;
         e += (long long)gridDim.x * 256) {
        int f = (int)(e % kpad);
        long long row = e / kpad;
        int i = (int)(row / KNN);
        float v = 0.f;
        if (f < din) {
            v = h[(long long)i * ldh + f];
        } else if (f < 2 * din) {
            int j = idx[row];
            int c = f - din;
            v = h[(long long)j * ldh + c] - h[(long long)i * ldh + c];
        }
        msg[e] = (_Float16)v;
    }
}

// ---------------------------------------------------------------------------
// Max over K neighbors -> z slice (both f32 for next gather and f16 for head)
// ---------------------------------------------------------------------------
__global__ __launch_bounds__(256)
void rowmax_kernel(const _Float16* __restrict__ h2, int dout,
                   float* __restrict__ zf, _Float16* __restrict__ zh,
                   int ldz, int N)
{
    long long tot = (long long)N * dout;
    for (long long e = (long long)blockIdx.x * 256 + threadIdx.x; e < tot;
         e += (long long)gridDim.x * 256) {
        int c = (int)(e % dout);
        int i = (int)(e / dout);
        const _Float16* p = h2 + (long long)i * KNN * dout + c;
        float m = -3.0e38f;
#pragma unroll 4
        for (int k = 0; k < KNN; ++k) m = fmaxf(m, (float)p[(long long)k * dout]);
        zf[(long long)i * ldz + c] = m;
        zh[(long long)i * ldz + c] = (_Float16)m;
    }
}

// ---------------------------------------------------------------------------
// Param prep: W (Kr x Nr f32) -> W^T f16 zero-padded to Np x Kp
// ---------------------------------------------------------------------------
__global__ __launch_bounds__(256)
void prep_wt_kernel(const float* __restrict__ W, _Float16* __restrict__ wtT,
                    int Kr, int Nr, int Kp, int Np)
{
    int tot = Kp * Np;
    for (int e = blockIdx.x * 256 + threadIdx.x; e < tot; e += gridDim.x * 256) {
        int k = e % Kp, n = e / Kp;
        float v = (k < Kr && n < Nr) ? W[(long long)k * Nr + n] : 0.f;
        wtT[(long long)n * Kp + k] = (_Float16)v;
    }
}

// Fold bias + BN(eval) into per-channel scale/shift; g==null => bias only
__global__ __launch_bounds__(256)
void prep_ss_kernel(const float* __restrict__ b,  const float* __restrict__ g,
                    const float* __restrict__ bt, const float* __restrict__ m,
                    const float* __restrict__ v,
                    float* __restrict__ scale, float* __restrict__ shift,
                    int d, int dp)
{
    int i = blockIdx.x * 256 + threadIdx.x;
    if (i >= dp) return;
    if (i >= d)      { scale[i] = 1.f; shift[i] = 0.f; }
    else if (g) {
        float s = g[i] * rsqrtf(v[i] + 1e-5f);
        scale[i] = s;
        shift[i] = (b[i] - m[i]) * s + bt[i];
    } else           { scale[i] = 1.f; shift[i] = b[i]; }
}

// ---------------------------------------------------------------------------
extern "C" void kernel_launch(void* const* d_in, const int* in_sizes, int n_in,
                              void* d_out, int out_size, void* d_ws, size_t ws_size,
                              hipStream_t stream)
{
    (void)n_in; (void)out_size; (void)ws_size;
    const int N  = in_sizes[0] / 13;          // 8192
    const long long ME = (long long)N * KNN;  // 163840 edges
    const float* x = (const float*)d_in[0];

    // ----- layer configuration -----
    const int dinH[5] = {13, 64, 64, 128, 256};   // input feature dims
    const int kp1[5]  = {32, 128, 128, 256, 512}; // padded msg widths (2*din)
    const int dco[5]  = {64, 64, 128, 256, 128};  // layer output dims
    const int zoff[5] = {0, 64, 128, 256, 512};   // column offsets into z (640)

    // ----- param pointers (dict flatten: per-conv W1,W2,b1,b2,bt1,bt2,g1,g2,m1,m2,v1,v2) -----
    const float* Wf[13]; int Kr[13], Nr[13], Kp[13], Np[13];
    const float *bv[13], *gv[13], *btv[13], *mv[13], *vv[13];
    for (int l = 0; l < 5; ++l) {
        int b0 = 1 + 12 * l;
        int a = 2 * l, b = 2 * l + 1;
        Wf[a] = (const float*)d_in[b0 + 0]; Kr[a] = 2 * dinH[l]; Nr[a] = dco[l]; Kp[a] = kp1[l]; Np[a] = dco[l];
        Wf[b] = (const float*)d_in[b0 + 1]; Kr[b] = dco[l];      Nr[b] = dco[l]; Kp[b] = dco[l]; Np[b] = dco[l];
        bv[a]  = (const float*)d_in[b0 + 2];  bv[b]  = (const float*)d_in[b0 + 3];
        btv[a] = (const float*)d_in[b0 + 4];  btv[b] = (const float*)d_in[b0 + 5];
        gv[a]  = (const float*)d_in[b0 + 6];  gv[b]  = (const float*)d_in[b0 + 7];
        mv[a]  = (const float*)d_in[b0 + 8];  mv[b]  = (const float*)d_in[b0 + 9];
        vv[a]  = (const float*)d_in[b0 + 10]; vv[b]  = (const float*)d_in[b0 + 11];
    }
    // head: (W,b) x3, no BN
    Wf[10] = (const float*)d_in[61]; Kr[10] = 640; Nr[10] = 512; Kp[10] = 640; Np[10] = 512;
    Wf[11] = (const float*)d_in[63]; Kr[11] = 512; Nr[11] = 256; Kp[11] = 512; Np[11] = 256;
    Wf[12] = (const float*)d_in[65]; Kr[12] = 256; Nr[12] = 4;   Kp[12] = 256; Np[12] = 64;
    bv[10] = (const float*)d_in[62]; bv[11] = (const float*)d_in[64]; bv[12] = (const float*)d_in[66];
    gv[10] = gv[11] = gv[12] = nullptr;
    btv[10] = btv[11] = btv[12] = nullptr;
    mv[10] = mv[11] = mv[12] = nullptr;
    vv[10] = vv[11] = vv[12] = nullptr;

    // ----- workspace carve-out -----
    char* base = (char*)d_ws;
    size_t off = 0;
    auto alloc = [&](size_t bytes) -> char* {
        char* p = base + off;
        off = (off + bytes + 255) & ~(size_t)255;
        return p;
    };
    float*     xT  = (float*)alloc((size_t)13 * N * 4);
    int*       idx = (int*)  alloc((size_t)N * KNN * 4);
    float*     zf  = (float*)alloc((size_t)N * 640 * 4);
    _Float16*  zh  = (_Float16*)alloc((size_t)N * 640 * 2);
    _Float16*  wt[13];
    float *ssS[13], *ssT[13];
    for (int i = 0; i < 13; ++i) {
        wt[i]  = (_Float16*)alloc((size_t)Kp[i] * Np[i] * 2);
        ssS[i] = (float*)alloc((size_t)Np[i] * 4);
        ssT[i] = (float*)alloc((size_t)Np[i] * 4);
    }
    _Float16* h1buf  = (_Float16*)alloc((size_t)ME * 256 * 2);  // gemm1 out / head t1
    _Float16* msgbuf = (_Float16*)alloc((size_t)ME * 512 * 2);  // msg / gemm2 out / head t2

    // ----- parameter prep (f16 transposed weights + folded BN scale/shift) -----
    for (int i = 0; i < 13; ++i) {
        int tot = Kp[i] * Np[i];
        prep_wt_kernel<<<(tot + 255) / 256, 256, 0, stream>>>(Wf[i], wt[i], Kr[i], Nr[i], Kp[i], Np[i]);
        prep_ss_kernel<<<(Np[i] + 255) / 256, 256, 0, stream>>>(bv[i], gv[i], btv[i], mv[i], vv[i],
                                                                ssS[i], ssT[i], Nr[i], Np[i]);
    }

    // ----- kNN -----
    xpose_kernel<<<(N * 13 + 255) / 256, 256, 0, stream>>>(x, xT, N);
    knn_topk_kernel<<<N, 128, 0, stream>>>(xT, x, idx, N);

    // ----- EdgeConv stack -----
    for (int l = 0; l < 5; ++l) {
        const float* hin = (l == 0) ? x : (zf + zoff[l - 1]);
        int ldh = (l == 0) ? 13 : 640;
        long long tot = ME * kp1[l];
        gather_msg_kernel<<<4096, 256, 0, stream>>>(hin, ldh, idx, msgbuf, dinH[l], kp1[l], tot);

        dim3 g((unsigned)(ME / BM), (unsigned)(dco[l] / BN));
        wmma_gemm_kernel<<<g, 256, 0, stream>>>(msgbuf, kp1[l], wt[2 * l], ssS[2 * l], ssT[2 * l],
                                                h1buf, dco[l], nullptr, 0, 0, kp1[l], 2);
        wmma_gemm_kernel<<<g, 256, 0, stream>>>(h1buf, dco[l], wt[2 * l + 1], ssS[2 * l + 1], ssT[2 * l + 1],
                                                msgbuf, dco[l], nullptr, 0, 0, dco[l], 2);
        rowmax_kernel<<<4096, 256, 0, stream>>>(msgbuf, dco[l], zf + zoff[l], zh + zoff[l], 640, N);
    }

    // ----- head MLP -----
    _Float16* t1 = h1buf;    // 8192 x 512 fits
    _Float16* t2 = msgbuf;   // 8192 x 256 fits
    dim3 gh1((unsigned)(N / BM), 512 / BN);
    wmma_gemm_kernel<<<gh1, 256, 0, stream>>>(zh, 640, wt[10], ssS[10], ssT[10],
                                              t1, 512, nullptr, 0, 0, 640, 1);
    dim3 gh2((unsigned)(N / BM), 256 / BN);
    wmma_gemm_kernel<<<gh2, 256, 0, stream>>>(t1, 512, wt[11], ssS[11], ssT[11],
                                              t2, 256, nullptr, 0, 0, 512, 1);
    dim3 gh3((unsigned)(N / BM), 1);
    wmma_gemm_kernel<<<gh3, 256, 0, stream>>>(t2, 256, wt[12], ssS[12], ssT[12],
                                              nullptr, 0, (float*)d_out, 4, 4, 256, 0);
}
